// DeformableVMFormer_59279138619678
// MI455X (gfx1250) — matrix operand order repaired and checked by reading, not compiled, not run on hardware
//
#include <hip/hip_runtime.h>

typedef __attribute__((ext_vector_type(16))) _Float16 v16h;
typedef __attribute__((ext_vector_type(8)))  float    v8f;

#define D_MODEL   256
#define N_HEADS   8
#define HEAD_DIM  32
#define N_LEVELS  4
#define N_POINTS  4
#define D_FFN     1024
#define NUM_Q     300
#define BATCH     2
#define S_TOTAL   5440
#define LN_EPS    1e-5f

// ---------------------------------------------------------------------------
// Pack fp32 weights into WMMA B-fragment layout (f16).
// Fragment order: ((kt*(N/16) + nt)*32 + lane)*16 + e
//   lane = sel*16 + r : element e = W[kt*32 + sel*16 + e][nt*16 + r]
// trans=1: src is [N,K] torch (out,in) layout -> reads are contiguous in e.
__global__ __launch_bounds__(256)
void pack_b(const float* __restrict__ src, _Float16* __restrict__ dst,
            int K, int N, int trans)
{
    int t = blockIdx.x * blockDim.x + threadIdx.x;
    int tilesN = N >> 4;
    int total = (K >> 5) * tilesN * 32;
    if (t >= total) return;
    int lane = t & 31;
    int nt = (t >> 5) % tilesN;
    int kt = t / (tilesN * 32);
    int r = lane & 15, sel = lane >> 4;
    int n = nt * 16 + r;
    int kb = kt * 32 + sel * 16;
    _Float16* d = dst + (size_t)t * 16;
#pragma unroll
    for (int e = 0; e < 16; ++e) {
        float v = trans ? src[(size_t)n * K + kb + e]
                        : src[(size_t)(kb + e) * N + n];
        d[e] = (_Float16)v;
    }
}

// ---------------------------------------------------------------------------
// WMMA GEMM: C[M,N] = act(A[M,K](f32) * B[K,N](packed f16) + bias)
// one wave -> 16x32 output (two 16x16 tiles, one shared A fragment).
// K % 32 == 0, N % 32 == 0.
__global__ __launch_bounds__(256)
void gemm_wmma(const float* __restrict__ A, const _Float16* __restrict__ Bp,
               const float* __restrict__ bias, float* __restrict__ C,
               int M, int N, int K, int relu)
{
    int wave  = (int)((blockIdx.x * (unsigned)blockDim.x + threadIdx.x) >> 5);
    int lane  = threadIdx.x & 31;
    int tilesN2 = N >> 5;
    int tilesM  = (M + 15) >> 4;
    int tM  = wave / tilesN2;
    int tN2 = wave - tM * tilesN2;
    if (tM >= tilesM) return;                      // wave-uniform exit

    int r = lane & 15, sel = lane >> 4;
    int mrow = tM * 16 + r; if (mrow >= M) mrow = M - 1;  // clamp: EXEC stays full
    int tilesN = N >> 4;

    const float* Arow = A + (size_t)mrow * K;
    const v16h*  B0 = (const v16h*)Bp + (size_t)(tN2 * 2) * 32 + lane;
    size_t bstep = (size_t)tilesN * 32;

    v8f acc0 = {}, acc1 = {};
    int nkt = K >> 5;
    for (int kt = 0; kt < nkt; ++kt) {
        const float4* Ap = (const float4*)(Arow + kt * 32 + sel * 16);
        __builtin_prefetch(Ap + 8, 0, 1);
        float4 p0 = Ap[0], p1 = Ap[1], p2 = Ap[2], p3 = Ap[3];
        v16h a;
        a[0]  = (_Float16)p0.x; a[1]  = (_Float16)p0.y;
        a[2]  = (_Float16)p0.z; a[3]  = (_Float16)p0.w;
        a[4]  = (_Float16)p1.x; a[5]  = (_Float16)p1.y;
        a[6]  = (_Float16)p1.z; a[7]  = (_Float16)p1.w;
        a[8]  = (_Float16)p2.x; a[9]  = (_Float16)p2.y;
        a[10] = (_Float16)p2.z; a[11] = (_Float16)p2.w;
        a[12] = (_Float16)p3.x; a[13] = (_Float16)p3.y;
        a[14] = (_Float16)p3.z; a[15] = (_Float16)p3.w;
        v16h b0 = B0[(size_t)kt * bstep];
        v16h b1 = B0[(size_t)kt * bstep + 32];
        acc0 = __builtin_amdgcn_wmma_f32_16x16x32_f16(false, a, false, b0,
                                                      (short)0, acc0, false, false);
        acc1 = __builtin_amdgcn_wmma_f32_16x16x32_f16(false, a, false, b1,
                                                      (short)0, acc1, false, false);
    }

    int col = tN2 * 32 + (lane & 15);
    float bv0 = bias[col], bv1 = bias[col + 16];
    int row0 = tM * 16 + sel * 8;
    float* Cb = C + (size_t)row0 * N + col;
    if (tM * 16 + 16 <= M) {                        // wave-uniform fast path
        if (relu) {
#pragma unroll
            for (int vr = 0; vr < 8; ++vr) {
                Cb[(size_t)vr * N]      = fmaxf(acc0[vr] + bv0, 0.f);
                Cb[(size_t)vr * N + 16] = fmaxf(acc1[vr] + bv1, 0.f);
            }
        } else {
#pragma unroll
            for (int vr = 0; vr < 8; ++vr) {
                Cb[(size_t)vr * N]      = acc0[vr] + bv0;
                Cb[(size_t)vr * N + 16] = acc1[vr] + bv1;
            }
        }
    } else {
#pragma unroll
        for (int vr = 0; vr < 8; ++vr) {
            if (row0 + vr < M) {
                float v0 = acc0[vr] + bv0, v1 = acc1[vr] + bv1;
                if (relu) { v0 = fmaxf(v0, 0.f); v1 = fmaxf(v1, 0.f); }
                Cb[(size_t)vr * N]      = v0;
                Cb[(size_t)vr * N + 16] = v1;
            }
        }
    }
}

// ---------------------------------------------------------------------------
// Fused residual-add + LayerNorm over 256 channels, one wave per row. O may == X.
__global__ __launch_bounds__(256)
void ln_fused(const float* __restrict__ X, const float* __restrict__ Y,
              const float* __restrict__ g, const float* __restrict__ b,
              float* __restrict__ O, int M)
{
    int wave = (int)((blockIdx.x * (unsigned)blockDim.x + threadIdx.x) >> 5);
    int lane = threadIdx.x & 31;
    if (wave >= M) return;
    const float* xr = X + (size_t)wave * D_MODEL;
    const float* yr = Y + (size_t)wave * D_MODEL;
    float v[8]; float s = 0.f;
#pragma unroll
    for (int i = 0; i < 8; ++i) {
        float t = xr[lane + i * 32] + yr[lane + i * 32];
        v[i] = t; s += t;
    }
    for (int o = 16; o; o >>= 1) s += __shfl_xor(s, o, 32);
    float mean = s * (1.f / 256.f);
    float vs = 0.f;
#pragma unroll
    for (int i = 0; i < 8; ++i) { float d = v[i] - mean; vs += d * d; }
    for (int o = 16; o; o >>= 1) vs += __shfl_xor(vs, o, 32);
    float rstd = rsqrtf(vs * (1.f / 256.f) + LN_EPS);
#pragma unroll
    for (int i = 0; i < 8; ++i) {
        int c = lane + i * 32;
        O[(size_t)wave * D_MODEL + c] = (v[i] - mean) * rstd * g[c] + b[c];
    }
}

// ---------------------------------------------------------------------------
// softmax over contiguous groups of 16 (MSDA attention weights), in place
__global__ __launch_bounds__(256)
void softmax16(float* __restrict__ A, int rows)
{
    int t = blockIdx.x * blockDim.x + threadIdx.x;
    if (t >= rows) return;
    float* p = A + (size_t)t * 16;
    float m = p[0];
#pragma unroll
    for (int i = 1; i < 16; ++i) m = fmaxf(m, p[i]);
    float e[16]; float s = 0.f;
#pragma unroll
    for (int i = 0; i < 16; ++i) { e[i] = __expf(p[i] - m); s += e[i]; }
    float inv = 1.f / s;
#pragma unroll
    for (int i = 0; i < 16; ++i) p[i] = e[i] * inv;
}

// softmax over rows of length n, one wave per row, in place
__global__ __launch_bounds__(256)
void softmax_rows(float* __restrict__ A, int rows, int n)
{
    int wave = (int)((blockIdx.x * (unsigned)blockDim.x + threadIdx.x) >> 5);
    int lane = threadIdx.x & 31;
    if (wave >= rows) return;
    float* p = A + (size_t)wave * n;
    float m = -3.4e38f;
    for (int j = lane; j < n; j += 32) m = fmaxf(m, p[j]);
    for (int o = 16; o; o >>= 1) m = fmaxf(m, __shfl_xor(m, o, 32));
    float s = 0.f;
    for (int j = lane; j < n; j += 32) { float e = __expf(p[j] - m); p[j] = e; s += e; }
    for (int o = 16; o; o >>= 1) s += __shfl_xor(s, o, 32);
    float inv = 1.f / s;
    for (int j = lane; j < n; j += 32) p[j] *= inv;
}

// ---------------------------------------------------------------------------
// Multi-scale deformable attention sampling.
__device__ __forceinline__ float msda_corner(const float* vb, int xi, int yi, int W, int H)
{
    if (xi < 0 || xi >= W || yi < 0 || yi >= H) return 0.f;
    return vb[(size_t)(yi * W + xi) * D_MODEL];
}

__global__ __launch_bounds__(256)
void msda_sample(const float* __restrict__ value, const float* __restrict__ off,
                 const float* __restrict__ attn, const float* __restrict__ ref,
                 float* __restrict__ out, int Lq)
{
    int tid = blockIdx.x * blockDim.x + threadIdx.x;
    int total = BATCH * Lq * D_MODEL;
    if (tid >= total) return;
    int d = tid & 31;
    int h = (tid >> 5) & 7;
    int q = (tid >> 8) % Lq;
    int b = tid / (Lq * D_MODEL);

    const int st[4] = {0, 4096, 5120, 5376};
    const int Hs[4] = {64, 32, 16, 8};

    size_t row = (size_t)b * Lq + q;
    float rx = ref[row * 2 + 0], ry = ref[row * 2 + 1];
    const float* offp = off  + row * 256 + h * 32;
    const float* ap   = attn + row * 128 + h * 16;

    float acc = 0.f;
#pragma unroll
    for (int l = 0; l < 4; ++l) {
        int H = Hs[l], W = Hs[l];
        const float* vb = value + ((size_t)b * S_TOTAL + st[l]) * D_MODEL + h * 32 + d;
#pragma unroll
        for (int p = 0; p < 4; ++p) {
            float ox = offp[(l * 4 + p) * 2 + 0];
            float oy = offp[(l * 4 + p) * 2 + 1];
            float w  = ap[l * 4 + p];
            float x = (rx + ox / (float)W) * W - 0.5f;
            float y = (ry + oy / (float)H) * H - 0.5f;
            float x0 = floorf(x), y0 = floorf(y);
            float fx = x - x0, fy = y - y0;
            int xi = (int)x0, yi = (int)y0;
            float v00 = msda_corner(vb, xi,     yi,     W, H);
            float v01 = msda_corner(vb, xi + 1, yi,     W, H);
            float v10 = msda_corner(vb, xi,     yi + 1, W, H);
            float v11 = msda_corner(vb, xi + 1, yi + 1, W, H);
            acc += w * (v00 * (1.f - fx) * (1.f - fy) + v01 * fx * (1.f - fy) +
                        v10 * (1.f - fx) * fy         + v11 * fx * fy);
        }
    }
    out[row * 256 + h * 32 + d] = acc;
}

// ---------------------------------------------------------------------------
// Decoder MHA logits via WMMA (single K=32 step per head)
__global__ __launch_bounds__(256)
void attn_logits(const float* __restrict__ Q, const float* __restrict__ Kt,
                 float* __restrict__ Lg, int L, float scale)
{
    int wave = (int)((blockIdx.x * (unsigned)blockDim.x + threadIdx.x) >> 5);
    int lane = threadIdx.x & 31;
    int tiles = (L + 15) >> 4;
    int per = tiles * tiles;
    int bh = wave / per;
    if (bh >= BATCH * N_HEADS) return;
    int tw = wave - bh * per;
    int tM = tw / tiles, tN = tw - tM * tiles;
    int b = bh >> 3, h = bh & 7;
    int r = lane & 15, sel = lane >> 4;
    int m = tM * 16 + r; if (m >= L) m = L - 1;
    int n = tN * 16 + r; if (n >= L) n = L - 1;

    const float4* qv = (const float4*)(Q  + ((size_t)(b * L + m)) * D_MODEL + h * 32 + sel * 16);
    const float4* kv = (const float4*)(Kt + ((size_t)(b * L + n)) * D_MODEL + h * 32 + sel * 16);
    float4 p0 = qv[0], p1 = qv[1], p2 = qv[2], p3 = qv[3];
    float4 k0 = kv[0], k1 = kv[1], k2 = kv[2], k3 = kv[3];
    v16h a, bb;
    a[0]  = (_Float16)p0.x; a[1]  = (_Float16)p0.y; a[2]  = (_Float16)p0.z; a[3]  = (_Float16)p0.w;
    a[4]  = (_Float16)p1.x; a[5]  = (_Float16)p1.y; a[6]  = (_Float16)p1.z; a[7]  = (_Float16)p1.w;
    a[8]  = (_Float16)p2.x; a[9]  = (_Float16)p2.y; a[10] = (_Float16)p2.z; a[11] = (_Float16)p2.w;
    a[12] = (_Float16)p3.x; a[13] = (_Float16)p3.y; a[14] = (_Float16)p3.z; a[15] = (_Float16)p3.w;
    bb[0]  = (_Float16)k0.x; bb[1]  = (_Float16)k0.y; bb[2]  = (_Float16)k0.z; bb[3]  = (_Float16)k0.w;
    bb[4]  = (_Float16)k1.x; bb[5]  = (_Float16)k1.y; bb[6]  = (_Float16)k1.z; bb[7]  = (_Float16)k1.w;
    bb[8]  = (_Float16)k2.x; bb[9]  = (_Float16)k2.y; bb[10] = (_Float16)k2.z; bb[11] = (_Float16)k2.w;
    bb[12] = (_Float16)k3.x; bb[13] = (_Float16)k3.y; bb[14] = (_Float16)k3.z; bb[15] = (_Float16)k3.w;
    v8f acc = {};
    acc = __builtin_amdgcn_wmma_f32_16x16x32_f16(false, a, false, bb,
                                                 (short)0, acc, false, false);
    int col = tN * 16 + (lane & 15);
#pragma unroll
    for (int vr = 0; vr < 8; ++vr) {
        int row = tM * 16 + sel * 8 + vr;
        if (row < L && col < L)
            Lg[((size_t)bh * L + row) * L + col] = acc[vr] * scale;
    }
}

// out[b,m,h*32+n] = sum_k P[bh,m,k] * V[b,k,h*32+n]   (K=L padded with zeros)
__global__ __launch_bounds__(256)
void attn_apply(const float* __restrict__ Pr, const float* __restrict__ V,
                float* __restrict__ O, int L)
{
    int wave = (int)((blockIdx.x * (unsigned)blockDim.x + threadIdx.x) >> 5);
    int lane = threadIdx.x & 31;
    int tilesM = (L + 15) >> 4;
    int per = tilesM * 2;
    int bh = wave / per;
    if (bh >= BATCH * N_HEADS) return;
    int tw = wave - bh * per;
    int tM = tw >> 1, tN = tw & 1;
    int b = bh >> 3, h = bh & 7;
    int r = lane & 15, sel = lane >> 4;
    int m = tM * 16 + r; if (m >= L) m = L - 1;
    int n = tN * 16 + r;

    const float* prow = Pr + ((size_t)bh * L + m) * L;
    v8f acc = {};
    int Kpad = (L + 31) & ~31;
    for (int k0 = 0; k0 < Kpad; k0 += 32) {
        v16h a, bb;
#pragma unroll
        for (int i = 0; i < 8; ++i) {
            int k1 = k0 + sel * 16 + i;
            int k2 = k0 + sel * 16 + 8 + i;
            a[i]     = (k1 < L) ? (_Float16)prow[k1] : (_Float16)0.f;
            a[8 + i] = (k2 < L) ? (_Float16)prow[k2] : (_Float16)0.f;
        }
#pragma unroll
        for (int e = 0; e < 16; ++e) {
            int kk = k0 + sel * 16 + e; if (kk >= L) kk = L - 1;   // A side is zero there
            bb[e] = (_Float16)V[((size_t)(b * L + kk)) * D_MODEL + h * 32 + n];
        }
        acc = __builtin_amdgcn_wmma_f32_16x16x32_f16(false, a, false, bb,
                                                     (short)0, acc, false, false);
    }
    int col = h * 32 + tN * 16 + (lane & 15);
#pragma unroll
    for (int vr = 0; vr < 8; ++vr) {
        int row = tM * 16 + sel * 8 + vr;
        if (row < L)
            O[((size_t)(b * L + row)) * D_MODEL + col] = acc[vr];
    }
}

// ---------------------------------------------------------------------------
// misc small kernels
__global__ __launch_bounds__(256)
void add2_k(float* __restrict__ o, const float* __restrict__ a,
            const float* __restrict__ b, long long n)
{
    long long t = blockIdx.x * 256LL + threadIdx.x;
    if (t < n) o[t] = a[t] + b[t];
}

__global__ __launch_bounds__(256)
void copy_k(float* __restrict__ o, const float* __restrict__ a, long long n)
{
    long long t = blockIdx.x * 256LL + threadIdx.x;
    if (t < n) o[t] = a[t];
}

__global__ __launch_bounds__(256)
void build_pos_level(const float* __restrict__ pos, const float* __restrict__ le,
                     float* __restrict__ out)
{
    long long t = blockIdx.x * 256LL + threadIdx.x;
    if (t >= (long long)BATCH * S_TOTAL * D_MODEL) return;
    int c = (int)(t & 255);
    int q = (int)((t >> 8) % S_TOTAL);
    int l = q < 4096 ? 0 : q < 5120 ? 1 : q < 5376 ? 2 : 3;
    out[t] = pos[t] + le[l * D_MODEL + c];
}

__global__ __launch_bounds__(256)
void enc_ref_k(float* __restrict__ refE)
{
    int q = blockIdx.x * blockDim.x + threadIdx.x;
    if (q >= S_TOTAL) return;
    int H, W, st;
    if (q < 4096)      { H = 64; W = 64; st = 0; }
    else if (q < 5120) { H = 32; W = 32; st = 4096; }
    else if (q < 5376) { H = 16; W = 16; st = 5120; }
    else               { H = 8;  W = 8;  st = 5376; }
    int qq = q - st; int i = qq / W, j = qq - i * W;
    float gx = (j + 0.5f) / W, gy = (i + 0.5f) / H;
    refE[(size_t)q * 2 + 0] = gx;
    refE[(size_t)q * 2 + 1] = gy;
    refE[((size_t)S_TOTAL + q) * 2 + 0] = gx;
    refE[((size_t)S_TOTAL + q) * 2 + 1] = gy;
}

__global__ __launch_bounds__(256)
void split_qe(const float* __restrict__ qe, float* __restrict__ qp,
              float* __restrict__ tgt)
{
    int t = blockIdx.x * blockDim.x + threadIdx.x;
    if (t >= NUM_Q * D_MODEL) return;
    int q = t >> 8, c = t & 255;
    float a = qe[(size_t)q * 512 + c];
    float b = qe[(size_t)q * 512 + 256 + c];
    qp[t] = a;  qp[NUM_Q * D_MODEL + t] = a;
    tgt[t] = b; tgt[NUM_Q * D_MODEL + t] = b;
}

__global__ __launch_bounds__(256)
void dec_ref_k(const float* __restrict__ qp, const float* __restrict__ rw,
               const float* __restrict__ rb, float* __restrict__ refD)
{
    int t = blockIdx.x * blockDim.x + threadIdx.x;
    if (t >= BATCH * NUM_Q) return;
    const float* x = qp + (size_t)t * D_MODEL;
    float s0 = rb[0], s1 = rb[1];
    for (int k = 0; k < D_MODEL; ++k) {
        float v = x[k];
        s0 += v * rw[k * 2 + 0];
        s1 += v * rw[k * 2 + 1];
    }
    refD[t * 2 + 0] = 1.f / (1.f + __expf(-s0));
    refD[t * 2 + 1] = 1.f / (1.f + __expf(-s1));
}

// ===========================================================================
extern "C" void kernel_launch(void* const* d_in, const int* in_sizes, int n_in,
                              void* d_out, int out_size, void* d_ws, size_t ws_size,
                              hipStream_t stream)
{
    (void)in_sizes; (void)n_in; (void)out_size; (void)ws_size;

    const float* src = (const float*)d_in[0];
    const float* pos = (const float*)d_in[1];
    const float* query_embed = (const float*)d_in[2];
    auto ENC = [&](int l, int j) { return (const float*)d_in[3 + l * 16 + j]; };
    auto DEC = [&](int l, int j) { return (const float*)d_in[99 + l * 22 + j]; };
    const float* level_embed = (const float*)d_in[231];
    const float* ref_w = (const float*)d_in[232];
    const float* ref_b = (const float*)d_in[233];
    // enc leaf order: 0 b1,1 b2,2 Wa,3 Wo,4 Wout,5 Wv,6 ba,7 bo,8 bout,9 bv,
    //                 10 ln1_b,11 ln1_g,12 ln2_b,13 ln2_g,14 w1,15 w2
    // dec leaf order: same 0..13, 14 ln3_b,15 ln3_g,16 in_b,17 in_w,18 out_b,
    //                 19 out_w,20 w1,21 w2

    _Float16* w16 = (_Float16*)d_ws;
    const size_t ENC_W16 = 753664;
    const size_t DEC_W16 = 1015808;
    auto encW = [&](int l) { return w16 + (size_t)l * ENC_W16; };
    auto decW = [&](int l) { return w16 + 6 * ENC_W16 + (size_t)l * DEC_W16; };
    const size_t O_WV = 0, O_WO = 65536, O_WA = 131072, O_WOUT = 163840,
                 O_W1 = 229376, O_W2 = 491520,
                 O_WQT = 753664, O_WKT = 819200, O_WVT = 884736, O_WOT = 950272;

    float* f = (float*)(w16 + 6 * ENC_W16 + 6 * DEC_W16);
    const size_t NSC = (size_t)BATCH * S_TOTAL * D_MODEL;
    const size_t NQC = (size_t)BATCH * NUM_Q * D_MODEL;
    float* posL    = f; f += NSC;
    float* mem     = f; f += NSC;
    float* bufQ    = f; f += NSC;
    float* bufVal  = f; f += NSC;
    float* bufOff  = f; f += NSC;
    float* bufAttn = f; f += NSC / 2;
    float* bufSamp = f; f += NSC;
    float* bufX    = f; f += NSC;
    float* bufFFN  = f; f += (size_t)BATCH * S_TOTAL * D_FFN;
    float* refE    = f; f += (size_t)BATCH * S_TOTAL * 2;
    float* qp      = f; f += NQC;
    float* tgt     = f; f += NQC;
    float* saq     = f; f += NQC;
    float* sak     = f; f += NQC;
    float* sav     = f; f += NQC;
    float* logits  = f; f += (size_t)BATCH * N_HEADS * NUM_Q * NUM_Q;
    float* refD    = f; f += (size_t)BATCH * NUM_Q * 2;

    auto gemm = [&](const float* A, const _Float16* Bw, const float* bias,
                    float* C, int M, int N, int K, int relu) {
        long long waves = (long long)((M + 15) / 16) * (N / 32);
        gemm_wmma<<<dim3((unsigned)((waves + 7) / 8)), 256, 0, stream>>>(
            A, Bw, bias, C, M, N, K, relu);
    };
    auto pack = [&](const float* s, _Float16* dst, int K, int N, int trans) {
        int total = (K >> 5) * (N >> 4) * 32;
        pack_b<<<(total + 255) / 256, 256, 0, stream>>>(s, dst, K, N, trans);
    };
    auto ln = [&](const float* X, const float* Y, const float* g, const float* b,
                  float* O, int M) {
        ln_fused<<<(M + 7) / 8, 256, 0, stream>>>(X, Y, g, b, O, M);
    };

    // ---- weight packing (fp32 -> f16 WMMA-fragment layout)
    for (int l = 0; l < 6; ++l) {
        _Float16* w = encW(l);
        pack(ENC(l, 5),  w + O_WV,   256, 256, 0);
        pack(ENC(l, 3),  w + O_WO,   256, 256, 0);
        pack(ENC(l, 2),  w + O_WA,   256, 128, 0);
        pack(ENC(l, 4),  w + O_WOUT, 256, 256, 0);
        pack(ENC(l, 14), w + O_W1,   256, 1024, 0);
        pack(ENC(l, 15), w + O_W2,   1024, 256, 0);
    }
    for (int l = 0; l < 6; ++l) {
        _Float16* w = decW(l);
        pack(DEC(l, 5),  w + O_WV,   256, 256, 0);
        pack(DEC(l, 3),  w + O_WO,   256, 256, 0);
        pack(DEC(l, 2),  w + O_WA,   256, 128, 0);
        pack(DEC(l, 4),  w + O_WOUT, 256, 256, 0);
        pack(DEC(l, 20), w + O_W1,   256, 1024, 0);
        pack(DEC(l, 21), w + O_W2,   1024, 256, 0);
        const float* inw = DEC(l, 17);           // [768,256] torch (out,in)
        pack(inw,          w + O_WQT, 256, 256, 1);
        pack(inw + 65536,  w + O_WKT, 256, 256, 1);
        pack(inw + 131072, w + O_WVT, 256, 256, 1);
        pack(DEC(l, 19),   w + O_WOT, 256, 256, 1);
    }

    build_pos_level<<<(unsigned)((NSC + 255) / 256), 256, 0, stream>>>(pos, level_embed, posL);
    enc_ref_k<<<(S_TOTAL + 255) / 256, 256, 0, stream>>>(refE);
    copy_k<<<(unsigned)((NSC + 255) / 256), 256, 0, stream>>>(mem, src, (long long)NSC);

    const int MS = BATCH * S_TOTAL;   // 10880
    const int MQ = BATCH * NUM_Q;     // 600

    // ---- encoder
    for (int l = 0; l < 6; ++l) {
        _Float16* w = encW(l);
        add2_k<<<(unsigned)((NSC + 255) / 256), 256, 0, stream>>>(bufQ, mem, posL, (long long)NSC);
        gemm(mem,  w + O_WV, ENC(l, 9), bufVal,  MS, 256, 256, 0);
        gemm(bufQ, w + O_WO, ENC(l, 7), bufOff,  MS, 256, 256, 0);
        gemm(bufQ, w + O_WA, ENC(l, 6), bufAttn, MS, 128, 256, 0);
        softmax16<<<(MS * N_HEADS + 255) / 256, 256, 0, stream>>>(bufAttn, MS * N_HEADS);
        msda_sample<<<(unsigned)((NSC + 255) / 256), 256, 0, stream>>>(
            bufVal, bufOff, bufAttn, refE, bufSamp, S_TOTAL);
        gemm(bufSamp, w + O_WOUT, ENC(l, 8), bufX, MS, 256, 256, 0);
        ln(mem, bufX, ENC(l, 11), ENC(l, 10), mem, MS);                 // ln1
        gemm(mem,    w + O_W1, ENC(l, 0), bufFFN, MS, 1024, 256, 1);
        gemm(bufFFN, w + O_W2, ENC(l, 1), bufX,   MS, 256, 1024, 0);
        ln(mem, bufX, ENC(l, 13), ENC(l, 12), mem, MS);                 // ln2
    }

    // ---- decoder preludes
    split_qe<<<(NUM_Q * D_MODEL + 255) / 256, 256, 0, stream>>>(query_embed, qp, tgt);
    dec_ref_k<<<(MQ + 255) / 256, 256, 0, stream>>>(qp, ref_w, ref_b, refD);

    const int tiles300 = (NUM_Q + 15) / 16;  // 19
    long long lgWaves = (long long)BATCH * N_HEADS * tiles300 * tiles300;
    long long apWaves = (long long)BATCH * N_HEADS * tiles300 * 2;

    // ---- decoder
    for (int l = 0; l < 6; ++l) {
        _Float16* w = decW(l);
        const float* in_b = DEC(l, 16);
        add2_k<<<(unsigned)((NQC + 255) / 256), 256, 0, stream>>>(bufQ, tgt, qp, (long long)NQC);
        gemm(bufQ, w + O_WQT, in_b,       saq, MQ, 256, 256, 0);
        gemm(bufQ, w + O_WKT, in_b + 256, sak, MQ, 256, 256, 0);
        gemm(tgt,  w + O_WVT, in_b + 512, sav, MQ, 256, 256, 0);
        attn_logits<<<dim3((unsigned)((lgWaves + 7) / 8)), 256, 0, stream>>>(
            saq, sak, logits, NUM_Q, 0.17677669529663687f /* 1/sqrt(32) */);
        softmax_rows<<<(BATCH * N_HEADS * NUM_Q + 7) / 8, 256, 0, stream>>>(
            logits, BATCH * N_HEADS * NUM_Q, NUM_Q);
        attn_apply<<<dim3((unsigned)((apWaves + 7) / 8)), 256, 0, stream>>>(
            logits, sav, bufSamp, NUM_Q);
        gemm(bufSamp, w + O_WOT, DEC(l, 18), bufX, MQ, 256, 256, 0);
        ln(tgt, bufX, DEC(l, 13), DEC(l, 12), tgt, MQ);                 // ln2
        add2_k<<<(unsigned)((NQC + 255) / 256), 256, 0, stream>>>(bufQ, tgt, qp, (long long)NQC);
        gemm(mem,  w + O_WV, DEC(l, 9), bufVal,  MS, 256, 256, 0);
        gemm(bufQ, w + O_WO, DEC(l, 7), bufOff,  MQ, 256, 256, 0);
        gemm(bufQ, w + O_WA, DEC(l, 6), bufAttn, MQ, 128, 256, 0);
        softmax16<<<(MQ * N_HEADS + 255) / 256, 256, 0, stream>>>(bufAttn, MQ * N_HEADS);
        msda_sample<<<(unsigned)((NQC + 255) / 256), 256, 0, stream>>>(
            bufVal, bufOff, bufAttn, refD, bufSamp, NUM_Q);
        gemm(bufSamp, w + O_WOUT, DEC(l, 8), bufX, MQ, 256, 256, 0);
        ln(tgt, bufX, DEC(l, 11), DEC(l, 10), tgt, MQ);                 // ln1
        gemm(tgt,    w + O_W1, DEC(l, 0), bufFFN, MQ, 1024, 256, 1);
        gemm(bufFFN, w + O_W2, DEC(l, 1), bufX,   MQ, 256, 1024, 0);
        ln(tgt, bufX, DEC(l, 15), DEC(l, 14), tgt, MQ);                 // ln3
    }

    copy_k<<<(unsigned)((NQC + 255) / 256), 256, 0, stream>>>((float*)d_out, tgt, (long long)NQC);
}
